// HybridAttentionBlock_19370302504977
// MI455X (gfx1250) — matrix operand order repaired and checked by reading, not compiled, not run on hardware
//
#include <hip/hip_runtime.h>
#include <hip/hip_bf16.h>
#include <math.h>

typedef __attribute__((ext_vector_type(16))) _Float16 v16h;
typedef __attribute__((ext_vector_type(8)))  float    v8f;

#define MTOK   73728      // B * H * W tokens
#define CCH    180
#define NHEAD  6
#define HD     30
#define NWIN   1152       // 8 * 12 * 12
#define HIDCH  720
#define DACH   96

__device__ __forceinline__ v8f wmma16(v16h a, v16h b, v8f c) {
  return __builtin_amdgcn_wmma_f32_16x16x32_f16(false, a, false, b, (short)0, c, false, false);
}

__device__ __forceinline__ float gelu_exact(float v) {
  return 0.5f * v * (1.f + erff(v * 0.70710678118654752f));
}

// window <-> flat token map (same map for partition gather and reverse scatter)
__device__ __forceinline__ int winmap(int m) {
  int w = m >> 6, n = m & 63;
  int b = w / 144, wi = w - b * 144;
  int wh = wi / 12, ww = wi - wh * 12;
  int r = n >> 3, c = n & 7;
  return b * 9216 + (wh * 8 + r) * 96 + (ww * 8 + c);
}

enum { GATHER_NONE = 0, GATHER_WIN = 1 };
enum { EPI_PLAIN = 0, EPI_GELU = 1, EPI_SCATTER_WIN = 2, EPI_RESID = 3 };

// ---------------------------------------------------------------------------
// Generic WMMA GEMM:  out[M][N] = A[M][K] (f32, f16-staged) @ Wt[N][K]^T + bias
// Block: 256 threads = 8 waves; tile 128(M) x 64(N); each wave: 16M x 64N
// (1 shared A fragment feeding 4 WMMAs per K-step of 32).
// ---------------------------------------------------------------------------
template <int GATHER, int EPI, bool AMUL>
__global__ __launch_bounds__(256) void k_gemm(
    const float* __restrict__ A, const float* __restrict__ Amul,
    const float* __restrict__ Wt, const float* __restrict__ bias,
    const float* __restrict__ resid, float* __restrict__ out,
    int M, int N, int K) {
  __shared__ _Float16 Ash[128][40];
  __shared__ _Float16 Bsh[32][72];

  const int blockM = blockIdx.x * 128;
  const int blockN = blockIdx.y * 64;
  const int tid  = threadIdx.x;
  const int wv   = tid >> 5;
  const int lane = tid & 31;
  const int lm   = lane & 15;
  const int lh   = lane >> 4;

  v8f acc0 = {}, acc1 = {}, acc2 = {}, acc3 = {};
  const int KT = (K + 31) / 32;

  for (int kt = 0; kt < KT; ++kt) {
    // stage A: 128 x 32 (f32 -> f16), optional row gather + elementwise mul;
    // prefetch next K-slice line per row (global_prefetch_b8).
    #pragma unroll
    for (int i = 0; i < 16; ++i) {
      int idx = i * 256 + tid;
      int row = idx >> 5, kk = idx & 31;
      int k = kt * 32 + kk;
      float v = 0.f;
      int grow = blockM + row;
      int srow = (GATHER == GATHER_WIN) ? winmap(grow) : grow;
      if (k < K) {
        v = A[(size_t)srow * K + k];
        if (AMUL) v *= Amul[(size_t)srow * K + k];
      }
      if (kk == 0 && k + 32 < K)
        __builtin_prefetch(&A[(size_t)srow * K + k + 32], 0, 0);
      Ash[row][kk] = (_Float16)v;
    }
    // stage B: 32(K) x 64(N), k-fast mapping -> coalesced 128B row reads
    #pragma unroll
    for (int i = 0; i < 8; ++i) {
      int idx = i * 256 + tid;
      int n = idx >> 5, kk = idx & 31;
      int k = kt * 32 + kk, col = blockN + n;
      float v = (k < K && col < N) ? Wt[(size_t)col * K + k] : 0.f;
      Bsh[kk][n] = (_Float16)v;
    }
    __syncthreads();

    v16h a, b;
    #pragma unroll
    for (int i = 0; i < 8; ++i) {
      a[i]     = Ash[wv * 16 + lm][lh * 16 + i];       // K = lh*16+i
      a[i + 8] = Ash[wv * 16 + lm][lh * 16 + 8 + i];   // K = lh*16+8+i
    }
    #pragma unroll
    for (int i = 0; i < 16; ++i) b[i] = Bsh[lh * 16 + i][lm];
    acc0 = wmma16(a, b, acc0);
    #pragma unroll
    for (int i = 0; i < 16; ++i) b[i] = Bsh[lh * 16 + i][16 + lm];
    acc1 = wmma16(a, b, acc1);
    #pragma unroll
    for (int i = 0; i < 16; ++i) b[i] = Bsh[lh * 16 + i][32 + lm];
    acc2 = wmma16(a, b, acc2);
    #pragma unroll
    for (int i = 0; i < 16; ++i) b[i] = Bsh[lh * 16 + i][48 + lm];
    acc3 = wmma16(a, b, acc3);
    __syncthreads();
  }

  // epilogue: 4 column sub-tiles; lane lm = col-in-tile; VGPR r -> row lh*8+r
  #pragma unroll
  for (int nt = 0; nt < 4; ++nt) {
    v8f acc = (nt == 0) ? acc0 : (nt == 1) ? acc1 : (nt == 2) ? acc2 : acc3;
    int col = blockN + nt * 16 + lm;
    if (col < N) {
      float bv = bias ? bias[col] : 0.f;
      #pragma unroll
      for (int r = 0; r < 8; ++r) {
        int mrow = blockM + wv * 16 + lh * 8 + r;
        float v = acc[r] + bv;
        if (EPI == EPI_GELU) v = gelu_exact(v);
        int orow = (EPI == EPI_SCATTER_WIN) ? winmap(mrow) : mrow;
        if (EPI == EPI_RESID) v += resid[(size_t)orow * N + col];
        out[(size_t)orow * N + col] = v;
      }
    }
  }
}

// ---------------------------------------------------------------------------
// CPB relative-position bias table + qkv bias vector (tiny, one block)
// ---------------------------------------------------------------------------
__global__ __launch_bounds__(256) void k_misc(
    const float* __restrict__ cpb_w1, const float* __restrict__ cpb_b1,
    const float* __restrict__ cpb_w2, const float* __restrict__ q_bias,
    const float* __restrict__ v_bias, float* __restrict__ rb,
    float* __restrict__ qkvb) {
  __shared__ float tbl[225][6];
  int t = threadIdx.x;
  for (int m = t; m < 225; m += 256) {
    int i = m / 15, j = m - 15 * (m / 15);
    float t0 = (float)(i - 7) / 7.f * 8.f;
    float t1 = (float)(j - 7) / 7.f * 8.f;
    t0 = copysignf(log2f(fabsf(t0) + 1.f), t0) * (1.f / 3.f);
    t1 = copysignf(log2f(fabsf(t1) + 1.f), t1) * (1.f / 3.f);
    float acc[6] = {0.f, 0.f, 0.f, 0.f, 0.f, 0.f};
    for (int k = 0; k < 512; ++k) {
      float h = fmaxf(t0 * cpb_w1[k * 2] + t1 * cpb_w1[k * 2 + 1] + cpb_b1[k], 0.f);
      #pragma unroll
      for (int hh = 0; hh < 6; ++hh) acc[hh] += h * cpb_w2[hh * 512 + k];
    }
    #pragma unroll
    for (int hh = 0; hh < 6; ++hh) tbl[m][hh] = acc[hh];
  }
  __syncthreads();
  for (int idx = t; idx < 6 * 4096; idx += 256) {
    int hh = idx >> 12, nm = idx & 4095;
    int n = nm >> 6, mm = nm & 63;
    int rpi = ((n >> 3) - (mm >> 3) + 7) * 15 + ((n & 7) - (mm & 7) + 7);
    rb[idx] = 16.f / (1.f + __expf(-tbl[rpi][hh]));
  }
  for (int j = t; j < 540; j += 256)
    qkvb[j] = (j < 180) ? q_bias[j] : (j < 360 ? 0.f : v_bias[j - 360]);
}

// ---------------------------------------------------------------------------
// Cosine attention per (window, head): 4 waves, full-WMMA S and O matmuls
// ---------------------------------------------------------------------------
__global__ __launch_bounds__(128) void k_attn(
    const float* __restrict__ qkv, const float* __restrict__ rb,
    const float* __restrict__ logit_scale, float* __restrict__ out) {
  __shared__ _Float16 Qs[64][34], Ks[64][34], Vs[64][34];
  __shared__ float Ss[64][66];
  __shared__ _Float16 Ps[64][68];

  int w = blockIdx.x / NHEAD, h = blockIdx.x - w * NHEAD;
  const float* base = qkv + (size_t)w * 64 * 540 + h * HD;
  float scale = __expf(fminf(logit_scale[h], 4.6051702f));
  int t = threadIdx.x;

  if (t < 64) {
    const float* qp = base + (size_t)t * 540;
    float tq[HD], tk[HD];
    float nq = 0.f, nk = 0.f;
    #pragma unroll
    for (int d = 0; d < HD; ++d) {
      tq[d] = qp[d];        nq += tq[d] * tq[d];
      tk[d] = qp[180 + d];  nk += tk[d] * tk[d];
    }
    nq = 1.f / fmaxf(sqrtf(nq), 1e-12f);
    nk = 1.f / fmaxf(sqrtf(nk), 1e-12f);
    #pragma unroll
    for (int d = 0; d < HD; ++d) {
      Qs[t][d] = (_Float16)(tq[d] * nq);
      Ks[t][d] = (_Float16)(tk[d] * nk);
      Vs[t][d] = (_Float16)qp[360 + d];
    }
    Qs[t][30] = Qs[t][31] = (_Float16)0.f;
    Ks[t][30] = Ks[t][31] = (_Float16)0.f;
    Vs[t][30] = Vs[t][31] = (_Float16)0.f;
  }
  __syncthreads();

  int wv = t >> 5, lane = t & 31, lm = lane & 15, lh = lane >> 4;

  // S = Qn @ Kn^T  (wave wv: rows 16wv..16wv+15, all 4 col tiles; K=32 padded)
  v16h a, b;
  #pragma unroll
  for (int i = 0; i < 8; ++i) {
    a[i]     = Qs[wv * 16 + lm][lh * 16 + i];
    a[i + 8] = Qs[wv * 16 + lm][lh * 16 + 8 + i];
  }
  for (int ct = 0; ct < 4; ++ct) {
    #pragma unroll
    for (int i = 0; i < 16; ++i) b[i] = Ks[ct * 16 + lm][lh * 16 + i];
    v8f c = {};
    c = wmma16(a, b, c);
    int col = ct * 16 + lm;
    #pragma unroll
    for (int r = 0; r < 8; ++r) {
      int row = wv * 16 + lh * 8 + r;
      Ss[row][col] = c[r] * scale + rb[h * 4096 + row * 64 + col];
    }
  }
  __syncthreads();

  if (t < 64) {
    float mx = -1e30f;
    for (int j = 0; j < 64; ++j) mx = fmaxf(mx, Ss[t][j]);
    float sum = 0.f;
    for (int j = 0; j < 64; ++j) { float e = __expf(Ss[t][j] - mx); Ss[t][j] = e; sum += e; }
    float inv = 1.f / sum;
    for (int j = 0; j < 64; ++j) Ps[t][j] = (_Float16)(Ss[t][j] * inv);
  }
  __syncthreads();

  // O = P @ V  (K=64 -> 2 k-steps; 2 col tiles, cols 30..31 are zero pad)
  for (int ct = 0; ct < 2; ++ct) {
    v8f c = {};
    for (int ks = 0; ks < 2; ++ks) {
      #pragma unroll
      for (int i = 0; i < 8; ++i) {
        a[i]     = Ps[wv * 16 + lm][ks * 32 + lh * 16 + i];
        a[i + 8] = Ps[wv * 16 + lm][ks * 32 + lh * 16 + 8 + i];
      }
      #pragma unroll
      for (int i = 0; i < 16; ++i) b[i] = Vs[ks * 32 + lh * 16 + i][ct * 16 + lm];
      c = wmma16(a, b, c);
    }
    int col = ct * 16 + lm;
    if (col < HD) {
      #pragma unroll
      for (int r = 0; r < 8; ++r) {
        int row = wv * 16 + lh * 8 + r;
        out[(size_t)(w * 64 + row) * CCH + h * HD + col] = c[r];
      }
    }
  }
}

// ---------------------------------------------------------------------------
// out = [resid +] LayerNorm(y) * g + b   (one wave per row of C floats)
// ---------------------------------------------------------------------------
template <bool HAS_RESID>
__global__ __launch_bounds__(256) void k_add_ln(
    const float* __restrict__ y, const float* __restrict__ resid,
    const float* __restrict__ g, const float* __restrict__ bta,
    float* __restrict__ out, int M, int C) {
  int row = blockIdx.x * 8 + (threadIdx.x >> 5);
  int lane = threadIdx.x & 31;
  if (row >= M) return;
  const float* yp = y + (size_t)row * C;
  float s = 0.f, s2 = 0.f;
  for (int j = lane; j < C; j += 32) { float v = yp[j]; s += v; s2 += v * v; }
  #pragma unroll
  for (int off = 16; off > 0; off >>= 1) {
    s  += __shfl_xor(s, off, 32);
    s2 += __shfl_xor(s2, off, 32);
  }
  float mean = s / (float)C;
  float var = s2 / (float)C - mean * mean;
  float rstd = rsqrtf(var + 1e-5f);
  float* op = out + (size_t)row * C;
  const float* rp = HAS_RESID ? resid + (size_t)row * C : nullptr;
  for (int j = lane; j < C; j += 32) {
    float v = (yp[j] - mean) * rstd * g[j] + bta[j];
    if (HAS_RESID) v += rp[j];
    op[j] = v;
  }
}

// ---------------------------------------------------------------------------
// Depthwise 5x5 conv, token-major [M][96] (coalesced across channels)
// ---------------------------------------------------------------------------
__global__ __launch_bounds__(96) void k_dw(
    const float* __restrict__ in, const float* __restrict__ wgt,
    const float* __restrict__ bias, float* __restrict__ out, int dil) {
  int p = blockIdx.x;
  int b = p / 9216, l = p - b * 9216;
  int y = l / 96, x = l - y * 96;
  int ch = threadIdx.x;
  float acc = bias[ch];
  #pragma unroll
  for (int dy = -2; dy <= 2; ++dy) {
    #pragma unroll
    for (int dx = -2; dx <= 2; ++dx) {
      int yy = y + dy * dil, xx = x + dx * dil;
      if (yy >= 0 && yy < 96 && xx >= 0 && xx < 96)
        acc += in[(size_t)(b * 9216 + yy * 96 + xx) * DACH + ch] *
               wgt[ch * 25 + (dy + 2) * 5 + (dx + 2)];
    }
  }
  out[(size_t)p * DACH + ch] = acc;
}

// ---------------------------------------------------------------------------
// Grouped 3x3 conv (groups=2), token-major [M][180]; LDS 3x3 neighborhood
// ---------------------------------------------------------------------------
__global__ __launch_bounds__(192) void k_convbody(
    const float* __restrict__ vn, const float* __restrict__ wgt,
    const float* __restrict__ bias, float* __restrict__ out) {
  __shared__ float tile[9][CCH];
  int p = blockIdx.x;
  int b = p / 9216, l = p - b * 9216;
  int y = l / 96, x = l - y * 96;
  int t = threadIdx.x;
  for (int i = t; i < 9 * CCH; i += 192) {
    int tap = i / CCH, c = i - tap * CCH;
    int dy = tap / 3 - 1, dx = tap % 3 - 1;
    int yy = y + dy, xx = x + dx;
    tile[tap][c] = (yy >= 0 && yy < 96 && xx >= 0 && xx < 96)
                       ? vn[(size_t)(b * 9216 + yy * 96 + xx) * CCH + c] : 0.f;
  }
  __syncthreads();
  if (t < CCH) {
    int g = t / 90;
    float acc = bias[t];
    const float* wp = wgt + (size_t)t * 90 * 9;  // w[co][ci][ky][kx]
    for (int ci = 0; ci < 90; ++ci) {
      #pragma unroll
      for (int tap = 0; tap < 9; ++tap)
        acc += tile[tap][g * 90 + ci] * wp[ci * 9 + tap];
    }
    out[(size_t)p * CCH + t] = acc;
  }
}

// ---------------------------------------------------------------------------
extern "C" void kernel_launch(void* const* d_in, const int* in_sizes, int n_in,
                              void* d_out, int out_size, void* d_ws, size_t ws_size,
                              hipStream_t stream) {
  const float* x           = (const float*)d_in[0];
  const float* qkv_w       = (const float*)d_in[1];
  const float* q_bias      = (const float*)d_in[2];
  const float* v_bias      = (const float*)d_in[3];
  const float* logit_scale = (const float*)d_in[4];
  const float* cpb_w1      = (const float*)d_in[5];
  const float* cpb_b1      = (const float*)d_in[6];
  const float* cpb_w2      = (const float*)d_in[7];
  const float* proj_w      = (const float*)d_in[8];
  const float* proj_b      = (const float*)d_in[9];
  const float* norm1_g     = (const float*)d_in[10];
  const float* norm1_b     = (const float*)d_in[11];
  const float* norm2_g     = (const float*)d_in[12];
  const float* norm2_b     = (const float*)d_in[13];
  const float* fc1_w       = (const float*)d_in[14];
  const float* fc1_b       = (const float*)d_in[15];
  const float* fc2_w       = (const float*)d_in[16];
  const float* fc2_b       = (const float*)d_in[17];
  const float* vab_p1_w    = (const float*)d_in[18];
  const float* vab_p1_b    = (const float*)d_in[19];
  const float* vab_pw_w    = (const float*)d_in[20];
  const float* vab_pw_b    = (const float*)d_in[21];
  const float* vab_dw_w    = (const float*)d_in[22];
  const float* vab_dw_b    = (const float*)d_in[23];
  const float* vab_dd_w    = (const float*)d_in[24];
  const float* vab_dd_b    = (const float*)d_in[25];
  const float* vab_p2_w    = (const float*)d_in[26];
  const float* vab_p2_b    = (const float*)d_in[27];
  const float* vab_ln_g    = (const float*)d_in[28];
  const float* vab_ln_b    = (const float*)d_in[29];
  const float* convbody_w  = (const float*)d_in[30];
  const float* convbody_b  = (const float*)d_in[31];
  const float* pe_w        = (const float*)d_in[32];
  const float* pe_b        = (const float*)d_in[33];

  const size_t M = MTOK;
  float* W0   = (float*)d_ws;
  float* RB   = W0;                       // 6*64*64
  float* QKVB = W0 + 24576;               // 540
  float* P    = W0 + 32768;               // M*540 : QKV, later CB|V2 ; HID = P..P+M*720
  float* OUT  = P   + M * 540;            // M*180 : attn out, later VN
  float* X1   = OUT + M * 180;            // M*180
  float* X2   = X1  + M * 180;            // M*180 : proj scratch, then x+pe
  float* V1   = X2  + M * 180;            // M*96  ; later fc2 out T (spans into AB)
  float* AB   = V1  + M * DACH;           // M*96
  float* AB2  = AB  + M * DACH;           // M*96
  float* CB   = P;
  float* V2   = P + M * 180;
  float* VN   = OUT;
  float* HID  = P;                        // M*720 == M*540 + M*180 (P + OUT)
  float* T    = V1;                       // M*180

  dim3 blk(256);
  const int MG = MTOK / 128;  // 576

  // bias table + qkv bias
  k_misc<<<1, 256, 0, stream>>>(cpb_w1, cpb_b1, cpb_w2, q_bias, v_bias, RB, QKVB);
  // QKV = window-gather(x) @ qkv_w^T + qkv_bias   (N=540 -> 9 tiles of 64)
  k_gemm<GATHER_WIN, EPI_PLAIN, false><<<dim3(MG, 9), blk, 0, stream>>>(
      x, nullptr, qkv_w, QKVB, nullptr, P, MTOK, 540, CCH);
  // windowed cosine attention
  k_attn<<<NWIN * NHEAD, 128, 0, stream>>>(P, RB, logit_scale, OUT);
  // proj + window reverse (scatter)
  k_gemm<GATHER_NONE, EPI_SCATTER_WIN, false><<<dim3(MG, 3), blk, 0, stream>>>(
      OUT, nullptr, proj_w, proj_b, nullptr, X2, MTOK, CCH, CCH);
  // x1 = x + LN(proj_out)
  k_add_ln<true><<<MTOK / 8, 256, 0, stream>>>(X2, x, norm1_g, norm1_b, X1, MTOK, CCH);
  // VAB: v1 = gelu(1x1 C->DA)
  k_gemm<GATHER_NONE, EPI_GELU, false><<<dim3(MG, 2), blk, 0, stream>>>(
      X1, nullptr, vab_p1_w, vab_p1_b, nullptr, V1, MTOK, DACH, CCH);
  // a = 1x1 DA->DA
  k_gemm<GATHER_NONE, EPI_PLAIN, false><<<dim3(MG, 2), blk, 0, stream>>>(
      V1, nullptr, vab_pw_w, vab_pw_b, nullptr, AB, MTOK, DACH, DACH);
  // depthwise 5x5 (dil 1), then 5x5 (dil 3)
  k_dw<<<MTOK, 96, 0, stream>>>(AB, vab_dw_w, vab_dw_b, AB2, 1);
  k_dw<<<MTOK, 96, 0, stream>>>(AB2, vab_dd_w, vab_dd_b, AB, 3);
  // v2 = 1x1 DA->C over (v1 * a) + x1
  k_gemm<GATHER_NONE, EPI_RESID, true><<<dim3(MG, 3), blk, 0, stream>>>(
      V1, AB, vab_p2_w, vab_p2_b, X1, V2, MTOK, CCH, DACH);
  // vn = LN(v2)
  k_add_ln<false><<<MTOK / 8, 256, 0, stream>>>(V2, nullptr, vab_ln_g, vab_ln_b, VN, MTOK, CCH);
  // cb = grouped 3x3
  k_convbody<<<MTOK, 192, 0, stream>>>(VN, convbody_w, convbody_b, CB);
  // x2 = x1 + pe(cb)  (1x1 C->C)
  k_gemm<GATHER_NONE, EPI_RESID, false><<<dim3(MG, 3), blk, 0, stream>>>(
      CB, nullptr, pe_w, pe_b, X1, X2, MTOK, CCH, CCH);
  // MLP
  k_gemm<GATHER_NONE, EPI_GELU, false><<<dim3(MG, 12), blk, 0, stream>>>(
      X2, nullptr, fc1_w, fc1_b, nullptr, HID, MTOK, HIDCH, CCH);
  k_gemm<GATHER_NONE, EPI_PLAIN, false><<<dim3(MG, 3), blk, 0, stream>>>(
      HID, nullptr, fc2_w, fc2_b, nullptr, T, MTOK, CCH, HIDCH);
  // out = x2 + LN(mlp_out)
  k_add_ln<true><<<MTOK / 8, 256, 0, stream>>>(T, X2, norm2_g, norm2_b, (float*)d_out, MTOK, CCH);
}
